// Generator_89000312308197
// MI455X (gfx1250) — compile-verified, matrix operand order checked by reading, not run on hardware
//
#include <hip/hip_runtime.h>
#include <cmath>

// ---------- types ----------
typedef __attribute__((ext_vector_type(16))) _Float16 v16h;
typedef __attribute__((ext_vector_type(8)))  _Float16 v8h;
typedef __attribute__((ext_vector_type(8)))  float    v8f;

// ---------- problem constants ----------
#define BATCH   2048
#define TSTEPS  32
#define DATA    16
#define HID     64
#define NOISE   8
#define CONDW   32
#define INITN   32
#define KPAD    128
#define ROWS    16        // batch rows per workgroup
#define NTHREADS 128      // 4 waves
#define NWAVES  4

// f16 weight offsets inside d_ws (element units)
#define OFF_INIT   0
#define OFF_DRIFT  57344
#define OFF_DIFF   114688
#define OFF_DIFFL4 163840
#define WTOT_F16   229376          // total f16 weight elements
#define BIAS_BYTE_OFF (WTOT_F16*2) // fp32 biases start here (458752, 16B aligned)

// per-MLP layer sub-offsets (f16 elements): each hidden layer block is [128][128]
#define L1_OFF 0
#define L2_OFF 16384
#define L3_OFF 32768
#define L4_OFF 49152

// ---------- shared memory layout (~281 KB, dynamic) ----------
struct Smem {
    _Float16 wA[57344];        // drift (or init) weights, transposed+padded
    _Float16 wB[49152];        // diffusion L1-L3
    _Float16 A0[ROWS * KPAD];  // MLP input (kept intact: shared by drift & diffusion)
    _Float16 A1[ROWS * KPAD];  // hidden ping
    _Float16 A2[ROWS * KPAD];  // hidden pong
    float    ybuf[ROWS * HID]; // y state
    float    ymid[ROWS * HID]; // midpoint state
    float    fbuf[ROWS * HID]; // drift output (tanh applied)
    float    gbuf[ROWS * 512]; // diffusion output (tanh applied)
    float    biasA[448];       // drift/init biases
    float    biasB[896];       // diffusion biases
    _Float16 condh[ROWS * CONDW];
    float    rW[HID * DATA];
    float    rb[DATA];
};

// ---------- weight prep: fp32 (K,N) -> f16 transposed [N][KPAD], zero-padded K ----------
__global__ void wprep_kernel(const float* __restrict__ W, int K, int N,
                             _Float16* __restrict__ dst) {
    int i = blockIdx.x * blockDim.x + threadIdx.x;
    if (i >= N * KPAD) return;
    int n = i >> 7;
    int k = i & (KPAD - 1);
    float v = (k < K) ? W[k * N + n] : 0.0f;
    dst[n * KPAD + k] = (_Float16)v;
}

// ---------- fast hardware-transcendental activations ----------
#define LOG2E_F 1.4426950408889634f
#define LN2_F   0.6931471805599453f

__device__ __forceinline__ float fast_exp(float x) {
#if defined(__AMDGCN__) && __has_builtin(__builtin_amdgcn_exp2f)
    return __builtin_amdgcn_exp2f(x * LOG2E_F);     // v_exp_f32
#else
    return __expf(x);
#endif
}

__device__ __forceinline__ float fast_log(float x) {
#if defined(__AMDGCN__) && __has_builtin(__builtin_amdgcn_logf)
    return __builtin_amdgcn_logf(x) * LN2_F;        // v_log_f32
#else
    return __logf(x);
#endif
}

__device__ __forceinline__ float fast_rcp(float x) {
#if defined(__AMDGCN__) && __has_builtin(__builtin_amdgcn_rcpf)
    return __builtin_amdgcn_rcpf(x);                // v_rcp_f32
#else
    return 1.0f / x;
#endif
}

// softplus(x) = max(x,0) + log1p(exp(-|x|))  (stable; ~6 VALU ops)
__device__ __forceinline__ float sp_act(float x) {
    return fmaxf(x, 0.0f) + fast_log(1.0f + fast_exp(-fabsf(x)));
}

// tanh via v_tanh_f32 if available, else exp2-based (~5 VALU ops)
__device__ __forceinline__ float th_act(float x) {
#if defined(__AMDGCN__) && __has_builtin(__builtin_amdgcn_tanhf)
    return __builtin_amdgcn_tanhf(x);               // v_tanh_f32
#else
    float e = fast_exp(2.0f * x);
    return 1.0f - 2.0f * fast_rcp(e + 1.0f);
#endif
}

// ---------- WMMA fragment loads ----------
__device__ __forceinline__ v16h ld_fragA(const _Float16* __restrict__ base,
                                         int lane, int kt) {
    int row = lane & 15;
    int k0  = kt * 32 + ((lane >> 4) << 3);
    v8h lo = *reinterpret_cast<const v8h*>(base + row * KPAD + k0);
    v8h hi = *reinterpret_cast<const v8h*>(base + row * KPAD + k0 + 16);
    return __builtin_shufflevector(lo, hi, 0,1,2,3,4,5,6,7,8,9,10,11,12,13,14,15);
}

__device__ __forceinline__ v16h ld_fragB(const _Float16* __restrict__ wt,
                                         int lane, int nt, int kt) {
    int row = nt * 16 + (lane & 15);
    int k0  = kt * 32 + ((lane >> 4) << 3);
    v8h lo = *reinterpret_cast<const v8h*>(wt + row * KPAD + k0);
    v8h hi = *reinterpret_cast<const v8h*>(wt + row * KPAD + k0 + 16);
    return __builtin_shufflevector(lo, hi, 0,1,2,3,4,5,6,7,8,9,10,11,12,13,14,15);
}

// One MLP layer: out(16 x N) = act( A(16x128) * WT[N][128]^T + bias )
// ACT: 0 = none, 1 = softplus, 2 = tanh (compile-time -> straight-line epilogue).
// F16OUT: true -> f16 row-major [16][KPAD]; false -> f32 [16][DSTSTRIDE].
template <int ACT, bool F16OUT, int NTILES, int DSTSTRIDE>
__device__ __forceinline__ void layer_forward(const _Float16* __restrict__ A,
                                              const _Float16* __restrict__ WT,
                                              const float* __restrict__ bias,
                                              _Float16* __restrict__ dstH,
                                              float* __restrict__ dstF) {
    const int wave = threadIdx.x >> 5;
    const int lane = threadIdx.x & 31;
#pragma unroll 1
    for (int nt = wave; nt < NTILES; nt += NWAVES) {
        const int ncol = nt * 16 + (lane & 15);
        float bv = bias[ncol];
        v8f c;
#pragma unroll
        for (int r = 0; r < 8; ++r) c[r] = bv;
#pragma unroll
        for (int kt = 0; kt < 4; ++kt) {
            v16h a = ld_fragA(A, lane, kt);
            v16h b = ld_fragB(WT, lane, nt, kt);
            c = __builtin_amdgcn_wmma_f32_16x16x32_f16(
                    false, a, false, b, (short)0, c, false, false);
        }
        const int mbase = (lane >> 4) << 3;
#pragma unroll
        for (int r = 0; r < 8; ++r) {
            int m = r + mbase;
            float v = c[r];
            if (ACT == 1)      v = sp_act(v);
            else if (ACT == 2) v = th_act(v);
            if (F16OUT) dstH[m * KPAD + ncol] = (_Float16)v;
            else        dstF[m * DSTSTRIDE + ncol] = v;
        }
    }
}

__device__ __forceinline__ void cpy16(void* dst, const void* src, int bytes) {
    uint4* d = (uint4*)dst;
    const uint4* s = (const uint4*)src;
    int n = bytes >> 4;
    for (int i = threadIdx.x; i < n; i += NTHREADS) d[i] = s[i];
}

// Build tx = [t, y(64), cond(32), 0-pad] as f16 into A0
__device__ __forceinline__ void build_A_tx(Smem& s, float tval,
                                           const float* __restrict__ ysrc) {
    _Float16 th = (_Float16)tval;
    for (int j = threadIdx.x; j < ROWS * KPAD; j += NTHREADS) {
        int r = j >> 7, c = j & (KPAD - 1);
        _Float16 v;
        if (c == 0)        v = th;
        else if (c < 65)   v = (_Float16)ysrc[r * HID + (c - 1)];
        else if (c < 97)   v = s.condh[r * CONDW + (c - 65)];
        else               v = (_Float16)0.0f;
        s.A0[j] = v;
    }
}

// 64->16 readout, plus ts column, written straight into d_out (B,T,17)
__device__ __forceinline__ void readout_write(Smem& s, float* __restrict__ out,
                                              int brow, int t, float tsv,
                                              const float* __restrict__ ysrc) {
    for (int j = threadIdx.x; j < ROWS * DATA; j += NTHREADS) {
        int r = j >> 4, c = j & 15;
        float acc = s.rb[c];
#pragma unroll 8
        for (int h = 0; h < HID; ++h) acc += ysrc[r * HID + h] * s.rW[h * DATA + c];
        out[((size_t)(brow + r) * TSTEPS + t) * (1 + DATA) + 1 + c] = acc;
    }
    if (threadIdx.x < ROWS)
        out[((size_t)(brow + threadIdx.x) * TSTEPS + t) * (1 + DATA)] = tsv;
}

// drift MLP: A0 -> fbuf (tanh)
__device__ __forceinline__ void run_drift(Smem& s) {
    layer_forward<1, true, 8, 0>(s.A0, s.wA + L1_OFF, s.biasA + 0,   s.A1, nullptr);
    __syncthreads();
    layer_forward<1, true, 8, 0>(s.A1, s.wA + L2_OFF, s.biasA + 128, s.A2, nullptr);
    __syncthreads();
    layer_forward<1, true, 8, 0>(s.A2, s.wA + L3_OFF, s.biasA + 256, s.A1, nullptr);
    __syncthreads();
    layer_forward<2, false, 4, HID>(s.A1, s.wA + L4_OFF, s.biasA + 384, nullptr, s.fbuf);
    __syncthreads();
}

// diffusion MLP: A0 -> gbuf (tanh), final layer weights streamed from global (L2-hot)
__device__ __forceinline__ void run_diffusion(Smem& s, const _Float16* __restrict__ wL4) {
    layer_forward<1, true, 8, 0>(s.A0, s.wB + L1_OFF, s.biasB + 0,   s.A1, nullptr);
    __syncthreads();
    layer_forward<1, true, 8, 0>(s.A1, s.wB + L2_OFF, s.biasB + 128, s.A2, nullptr);
    __syncthreads();
    layer_forward<1, true, 8, 0>(s.A2, s.wB + L3_OFF, s.biasB + 256, s.A1, nullptr);
    __syncthreads();
    layer_forward<2, false, 32, 512>(s.A1, wL4, s.biasB + 384, nullptr, s.gbuf);
    __syncthreads();
}

// y_next[j] = ybuf[j] + cf*fbuf[j] + cg*(g . Ik) ; Ik = dW[k]*sqrt(dt)
__device__ __forceinline__ void combine_step(Smem& s, const float* __restrict__ dWg,
                                             int k, int brow, float sdt,
                                             float cf, float cg,
                                             float* __restrict__ dst) {
    const int j0 = threadIdx.x * 8;       // 1024 jobs / 128 threads; 8 js share one row r
    const int r  = j0 >> 6;
    const float* dwp = dWg + ((size_t)k * BATCH + brow + r) * NOISE;
    float Ik[NOISE];
#pragma unroll
    for (int n = 0; n < NOISE; ++n) Ik[n] = dwp[n] * sdt;
#pragma unroll
    for (int j = j0; j < j0 + 8; ++j) {
        int h = j & 63;
        float gy = 0.0f;
#pragma unroll
        for (int n = 0; n < NOISE; ++n)
            gy += s.gbuf[r * 512 + h * NOISE + n] * Ik[n];
        dst[j] = s.ybuf[j] + cf * s.fbuf[j] + cg * gy;
    }
}

// ---------- main persistent SDE kernel: one WG owns 16 batch rows ----------
__global__ __launch_bounds__(NTHREADS, 1)
void sde_kernel(const float* __restrict__ ts,
                const float* __restrict__ cond,
                const float* __restrict__ initn,
                const float* __restrict__ dWg,
                const _Float16* __restrict__ wbase,
                const float* __restrict__ bbase,
                const float* __restrict__ rW,
                const float* __restrict__ rb,
                float* __restrict__ out) {
    extern __shared__ char smem_raw[];
    Smem& s = *reinterpret_cast<Smem*>(smem_raw);
    const int brow = blockIdx.x * ROWS;
    const int tid  = threadIdx.x;

    // ---- stage constants: readout, cond tile ----
    cpy16(s.rW, rW, HID * DATA * 4);
    if (tid < DATA) s.rb[tid] = rb[tid];
    for (int j = tid; j < ROWS * CONDW; j += NTHREADS) {
        int r = j >> 5, c = j & 31;
        s.condh[j] = (_Float16)cond[(size_t)(brow + r) * CONDW + c];
    }

    // warm L2 with the global-resident diffusion-L4 weights (global_prefetch_b8)
    {
        const char* p = (const char*)(wbase + OFF_DIFFL4);
        for (int off = tid * 128; off < 512 * KPAD * 2; off += NTHREADS * 128)
            __builtin_prefetch(p + off, 0, 1);
    }

    // ---- phase 0: initial-condition MLP (reuses drift LDS slots) ----
    cpy16(s.wA, wbase + OFF_INIT, 57344 * 2);
    cpy16(s.biasA, bbase + 0, 448 * 4);
    for (int j = tid; j < ROWS * KPAD; j += NTHREADS) {
        int r = j >> 7, c = j & (KPAD - 1);
        _Float16 v;
        if (c < INITN)            v = (_Float16)initn[(size_t)(brow + r) * INITN + c];
        else if (c < INITN+CONDW) v = (_Float16)cond[(size_t)(brow + r) * CONDW + (c - INITN)];
        else                      v = (_Float16)0.0f;
        s.A0[j] = v;
    }
    __syncthreads();
    layer_forward<1, true, 8, 0>(s.A0, s.wA + L1_OFF, s.biasA + 0,   s.A1, nullptr);
    __syncthreads();
    layer_forward<1, true, 8, 0>(s.A1, s.wA + L2_OFF, s.biasA + 128, s.A2, nullptr);
    __syncthreads();
    layer_forward<1, true, 8, 0>(s.A2, s.wA + L3_OFF, s.biasA + 256, s.A1, nullptr);
    __syncthreads();
    layer_forward<0, false, 4, HID>(s.A1, s.wA + L4_OFF, s.biasA + 384, nullptr, s.ybuf);
    __syncthreads();
    readout_write(s, out, brow, 0, ts[0], s.ybuf);

    // ---- phase 1: load drift + diffusion weights ----
    cpy16(s.wA, wbase + OFF_DRIFT, 57344 * 2);
    cpy16(s.wB, wbase + OFF_DIFF,  49152 * 2);
    cpy16(s.biasA, bbase + 448, 448 * 4);
    cpy16(s.biasB, bbase + 896, 896 * 4);
    const _Float16* wDiffL4 = wbase + OFF_DIFFL4;
    __syncthreads();

    // ---- midpoint SDE scan ----
    for (int k = 0; k < TSTEPS - 1; ++k) {
        float t0 = ts[k], t1 = ts[k + 1];
        float dt  = t1 - t0;
        float sdt = sqrtf(dt);
        float tm  = t0 + 0.5f * dt;

        // substep A: evaluate at (t0, y) -> y_mid
        build_A_tx(s, t0, s.ybuf);
        __syncthreads();
        run_drift(s);
        run_diffusion(s, wDiffL4);
        combine_step(s, dWg, k, brow, sdt, 0.5f * dt, 0.5f, s.ymid);
        __syncthreads();

        // substep B: evaluate at (tm, y_mid) -> y1 (overwrites ybuf elementwise)
        build_A_tx(s, tm, s.ymid);
        __syncthreads();
        run_drift(s);
        run_diffusion(s, wDiffL4);
        combine_step(s, dWg, k, brow, sdt, dt, 1.0f, s.ybuf);
        __syncthreads();

        readout_write(s, out, brow, k + 1, t1, s.ybuf);
        __syncthreads();
    }
}

// ---------- host side ----------
extern "C" void kernel_launch(void* const* d_in, const int* in_sizes, int n_in,
                              void* d_out, int out_size, void* d_ws, size_t ws_size,
                              hipStream_t stream) {
    (void)in_sizes; (void)n_in; (void)out_size; (void)ws_size;
    // dict/pytree flatten order:
    // 0 series, 1 ts, 2 encoder_forward, 3 init_noise, 4 dW,
    // 5..12 init (W,b)x4, 13..20 drift, 21..28 diffusion, 29 readout_W, 30 readout_b
    const float* ts    = (const float*)d_in[1];
    const float* cond  = (const float*)d_in[2];
    const float* initn = (const float*)d_in[3];
    const float* dW    = (const float*)d_in[4];
    const float* rW    = (const float*)d_in[29];
    const float* rb    = (const float*)d_in[30];

    _Float16* wf = (_Float16*)d_ws;
    float*    bf = (float*)((char*)d_ws + BIAS_BYTE_OFF);
    float*    out = (float*)d_out;

    // weight transposes (fp32 -> padded transposed f16)
    struct WSpec { int idx, K, N, off; };
    const WSpec wspec[12] = {
        {5,  64, 128, OFF_INIT  + L1_OFF}, {7, 128, 128, OFF_INIT  + L2_OFF},
        {9, 128, 128, OFF_INIT  + L3_OFF}, {11,128,  64, OFF_INIT  + L4_OFF},
        {13, 97, 128, OFF_DRIFT + L1_OFF}, {15,128, 128, OFF_DRIFT + L2_OFF},
        {17,128, 128, OFF_DRIFT + L3_OFF}, {19,128,  64, OFF_DRIFT + L4_OFF},
        {21, 97, 128, OFF_DIFF  + L1_OFF}, {23,128, 128, OFF_DIFF  + L2_OFF},
        {25,128, 128, OFF_DIFF  + L3_OFF}, {27,128, 512, OFF_DIFFL4},
    };
    for (int i = 0; i < 12; ++i) {
        int elems = wspec[i].N * KPAD;
        wprep_kernel<<<(elems + 255) / 256, 256, 0, stream>>>(
            (const float*)d_in[wspec[i].idx], wspec[i].K, wspec[i].N, wf + wspec[i].off);
    }

    // biases: contiguous fp32 region [init 448][drift 448][diff 896]
    struct BSpec { int idx, off, n; };
    const BSpec bspec[12] = {
        {6,   0,128},{8, 128,128},{10,256,128},{12,384, 64},
        {14,448,128},{16,576,128},{18,704,128},{20,832, 64},
        {22,896,128},{24,1024,128},{26,1152,128},{28,1280,512},
    };
    for (int i = 0; i < 12; ++i)
        hipMemcpyAsync(bf + bspec[i].off, d_in[bspec[i].idx],
                       (size_t)bspec[i].n * sizeof(float),
                       hipMemcpyDeviceToDevice, stream);

    sde_kernel<<<BATCH / ROWS, NTHREADS, sizeof(Smem), stream>>>(
        ts, cond, initn, dW, wf, bf, rW, rb, out);
}
